// Multi_EfficientAttention_67473936220438
// MI455X (gfx1250) — compile-verified, matrix operand order checked by reading, not run on hardware
//
#include <hip/hip_runtime.h>
#include <hip/hip_bf16.h>

// ---------------------------------------------------------------------------
// Types for CDNA5 WMMA (wave32): v_wmma_f32_16x16x32_bf16
// ---------------------------------------------------------------------------
typedef __attribute__((ext_vector_type(16))) __bf16 v16bf;
typedef __attribute__((ext_vector_type(2)))  __bf16 v2bf;
typedef __attribute__((ext_vector_type(8)))  float  v8f;

static __device__ __forceinline__ __bf16 f2bf(float f) {
    // round-to-nearest-even fp32 -> bf16
    unsigned u = __builtin_bit_cast(unsigned, f);
    u += 0x7FFFu + ((u >> 16) & 1u);
    unsigned short h = (unsigned short)(u >> 16);
    return __builtin_bit_cast(__bf16, h);
}

static __device__ __forceinline__ v8f wmma_bf16(v16bf a, v16bf b, v8f c) {
    return __builtin_amdgcn_wmma_f32_16x16x32_bf16(
        /*neg_a=*/false, a, /*neg_b=*/false, b,
        /*c_mod=*/(short)0, c, /*reuse_a=*/false, /*reuse_b=*/false);
}

// K-index within a 16x32 (A) or 32x16 (B) bf16 fragment, per ISA 7.12.2:
//   lane[4] selects the 8-wide K half, VGPR j in 0..3 -> K 0..7(+half),
//   VGPR j in 4..7 -> K 16..23(+half); each VGPR packs (k, k+1).
static __device__ __forceinline__ int frag_kbase(int j, int lane) {
    return ((j >= 4) ? 16 : 0) + ((lane >> 4) << 3) + ((j & 3) << 1);
}

// ---------------------------------------------------------------------------
// CDNA5 async global->LDS copy (tracked by ASYNCcnt). Per ISA 10.x/15.18:
//   each lane copies 16 bytes from its global address to its LDS offset.
// LDS offset = low 32 bits of the flat shared-aperture address (ISA 10.2).
// ---------------------------------------------------------------------------
static __device__ __forceinline__ void async_load_b128(unsigned lds_off,
                                                       const void* gaddr) {
    asm volatile("global_load_async_to_lds_b128 %0, %1, off"
                 :: "v"(lds_off), "v"(gaddr)
                 : "memory");
}

// ---------------------------------------------------------------------------
// GEMM + bias (+optional ReLU):  C[M,N] = act(A[M,K] @ W[K,N] + bias)
//   A: fp32 (converted) or bf16 (direct);  W: fp32, converted to bf16 in LDS
//   block = 256 threads (8 waves); wave w -> rows m0..m0+15, block cols 64
// ---------------------------------------------------------------------------
template <bool A_IS_BF16, bool OUT_BF16, bool RELU>
__global__ __launch_bounds__(256)
void gemm_bias_kernel(const void* __restrict__ Aptr,
                      const float* __restrict__ W,
                      const float* __restrict__ bias,
                      void* __restrict__ Cptr,
                      int M, int N, int K) {
    const int lane = threadIdx.x & 31;
    const int wave = threadIdx.x >> 5;
    const int m0 = blockIdx.x * 128 + wave * 16;
    const int n0 = blockIdx.y * 64;

    // W tile staged as bf16, laid out [n][k] so k-pairs are contiguous.
    __shared__ __bf16 Bs[64][40];   // 64 n-rows x (32 k + 8 pad)

    v8f acc[4] = {};

    const int mrow = m0 + (lane & 15);

    for (int k0 = 0; k0 < K; k0 += 32) {
        // ---- cooperative load of 32x64 W tile -> bf16 LDS (coalesced on n)
        for (int idx = threadIdx.x; idx < 32 * 64; idx += 256) {
            int kk = idx >> 6;          // 0..31
            int nn = idx & 63;          // 0..63
            Bs[nn][kk] = f2bf(W[(size_t)(k0 + kk) * N + (n0 + nn)]);
        }
        // ---- prefetch next W tile into L2 while we compute on this one
        if (k0 + 32 < K) {
            const float* nw = W + (size_t)(k0 + 32 + (threadIdx.x >> 3)) * N
                                + n0 + (threadIdx.x & 7) * 8;
            __builtin_prefetch(nw, 0, 1);
        }
        __syncthreads();

        // ---- A fragment (16x32), one per wave, direct from global
        v16bf a;
        if (A_IS_BF16) {
            const __bf16* Ab = (const __bf16*)Aptr + (size_t)mrow * K + k0;
#pragma unroll
            for (int j = 0; j < 8; ++j) {
                int kk = frag_kbase(j, lane);
                a[2 * j]     = Ab[kk];
                a[2 * j + 1] = Ab[kk + 1];
            }
        } else {
            const float* Af = (const float*)Aptr + (size_t)mrow * K + k0;
#pragma unroll
            for (int j = 0; j < 8; ++j) {
                int kk = frag_kbase(j, lane);
                a[2 * j]     = f2bf(Af[kk]);
                a[2 * j + 1] = f2bf(Af[kk + 1]);
            }
        }

        // ---- 4 B fragments from LDS, 4 WMMAs (A reused)
#pragma unroll
        for (int f = 0; f < 4; ++f) {
            v16bf b;
            int nn = f * 16 + (lane & 15);
#pragma unroll
            for (int j = 0; j < 8; ++j) {
                int kk = frag_kbase(j, lane);
                b[2 * j]     = Bs[nn][kk];
                b[2 * j + 1] = Bs[nn][kk + 1];
            }
            acc[f] = wmma_bf16(a, b, acc[f]);
        }
        __syncthreads();
    }

    // ---- epilogue: bias (+relu), store per C/D layout (VGPR r -> row r/r+8)
#pragma unroll
    for (int f = 0; f < 4; ++f) {
        int n = n0 + f * 16 + (lane & 15);
        float bv = bias[n];
#pragma unroll
        for (int r = 0; r < 8; ++r) {
            int row = m0 + r + ((lane >> 4) << 3);
            float val = acc[f][r] + bv;
            if (RELU) val = fmaxf(val, 0.0f);
            if (OUT_BF16)
                ((__bf16*)Cptr)[(size_t)row * N + n] = f2bf(val);
            else
                ((float*)Cptr)[(size_t)row * N + n] = val;
        }
    }
}

// ---------------------------------------------------------------------------
// Flash attention (online softmax).
//   Block = 128 threads (4 waves) sharing one (batch, head); each wave owns
//   16 query rows. K chunks are double-buffered in LDS and fetched with
//   global_load_async_to_lds_b128 one chunk ahead (ASYNCcnt pipelining,
//   loads complete in order -> s_wait_asynccnt 2 retires the older chunk).
//   V is staged transposed ([dv][key]) via VGPRs so B-fragment key-pairs
//   are contiguous ds_load_b32.
//   q,k,v bf16 [B*N, 1024] row-major; head h occupies cols h*64..h*64+63.
//   Scores scaled by 1/sqrt(Nk) per the reference.
// ---------------------------------------------------------------------------
__global__ __launch_bounds__(128)
void flash_attn_kernel(const __bf16* __restrict__ q,
                       const __bf16* __restrict__ k,
                       const __bf16* __restrict__ v,
                       __bf16* __restrict__ o,
                       int Nq, int Nk, float scale) {
    const int LD = 1024;
    const int tid = threadIdx.x;
    const int lane = tid & 31;
    const int wave = tid >> 5;
    const int b = blockIdx.z;
    const int h = blockIdx.y;
    const int n0 = blockIdx.x * 64 + wave * 16;
    const int kh = (lane >> 4) << 3;

    // K: double-buffered, row stride 72 elems = 144 B (16B-aligned for async
    // B128 writes; 36-bank stride -> conflict-free 16-row fragment reads).
    __shared__ __bf16 Ks[2][32][72];     // [buf][key][d] (d-pairs contiguous)
    __shared__ __bf16 Vt[64][34];        // [dv][key]     (key-pairs contiguous)
    __shared__ __bf16 Plds[4][16][36];   // per-wave P transpose buffer

    // ---- issue async K-chunk copy: 32 keys x 128 B, 2 B128 ops per thread
    auto issue_k_chunk = [&](int kcn, int nbuf) {
        for (int idx = tid; idx < 256; idx += 128) {
            int key = idx >> 3;          // 0..31
            int c   = idx & 7;           // 16-byte chunk within the row
            const __bf16* src =
                k + (size_t)(b * Nk + kcn + key) * LD + h * 64 + c * 8;
            async_load_b128((unsigned)(size_t)&Ks[nbuf][key][c * 8], src);
        }
    };

    // Q fragments: rows n0..n0+15, d = 0..63 as two 16x32 A-frags (kept live)
    v16bf qf[2];
    {
        const __bf16* qrow =
            q + (size_t)(b * Nq + n0 + (lane & 15)) * LD + h * 64;
#pragma unroll
        for (int c = 0; c < 2; ++c)
#pragma unroll
            for (int j = 0; j < 8; ++j) {
                int d = c * 32 + frag_kbase(j, lane);
                qf[c][2 * j]     = qrow[d];
                qf[c][2 * j + 1] = qrow[d + 1];
            }
    }

    float mrow[8], lrow[8];
#pragma unroll
    for (int r = 0; r < 8; ++r) { mrow[r] = -3.0e38f; lrow[r] = 0.0f; }
    v8f oacc[4] = {};

    // prologue: start the DMA for chunk 0
    issue_k_chunk(0, 0);

    for (int kc = 0; kc < Nk; kc += 32) {
        const int buf = (kc >> 5) & 1;
        const bool have_next = (kc + 32) < Nk;

        // ---- kick off next K chunk's DMA into the other buffer
        if (have_next) issue_k_chunk(kc + 32, buf ^ 1);

        // ---- stage V chunk transposed (coalesced global, VGPR path)
        for (int idx = tid; idx < 32 * 32; idx += 128) {
            int key = idx >> 5;          // 0..31
            int du  = idx & 31;          // uint index: channels 2*du, 2*du+1
            const __bf16* vrow = v + (size_t)(b * Nk + kc + key) * LD + h * 64;
            unsigned vw = *(const unsigned*)(vrow + 2 * du);
            v2bf vp = __builtin_bit_cast(v2bf, vw);
            Vt[2 * du][key]     = vp[0];
            Vt[2 * du + 1][key] = vp[1];
        }
        // ---- prefetch next chunk's V rows into L2
        if (have_next) {
            const __bf16* nv =
                v + (size_t)(b * Nk + kc + 32 + lane) * LD + h * 64;
            __builtin_prefetch(nv, 0, 1);
        }

        // ---- retire this chunk's async K copies (in-order completion:
        //      <=2 outstanding means the older chunk's 2 ops are done)
        if (have_next) asm volatile("s_wait_asynccnt 0x2" ::: "memory");
        else           asm volatile("s_wait_asynccnt 0x0" ::: "memory");
        __syncthreads();

        // ---- S = Q @ K^T for a 16x32 score strip (two 16x16 D tiles)
        v8f s0 = {}, s1 = {};
#pragma unroll
        for (int c = 0; c < 2; ++c) {
            v16bf kf0, kf1;
            int kr0 = lane & 15;
            int kr1 = 16 + (lane & 15);
#pragma unroll
            for (int j = 0; j < 8; ++j) {
                int d = c * 32 + frag_kbase(j, lane);
                kf0[2 * j]     = Ks[buf][kr0][d];
                kf0[2 * j + 1] = Ks[buf][kr0][d + 1];
                kf1[2 * j]     = Ks[buf][kr1][d];
                kf1[2 * j + 1] = Ks[buf][kr1][d + 1];
            }
            s0 = wmma_bf16(qf[c], kf0, s0);
            s1 = wmma_bf16(qf[c], kf1, s1);
        }

        // ---- online softmax update (row-wise over 32 keys)
        float alpha[8];
#pragma unroll
        for (int r = 0; r < 8; ++r) {
            float a0 = s0[r] * scale;
            float a1 = s1[r] * scale;
            float mx = fmaxf(a0, a1);
            // reduce over the 16-lane half-wave holding this row
            mx = fmaxf(mx, __shfl_xor(mx, 1, 32));
            mx = fmaxf(mx, __shfl_xor(mx, 2, 32));
            mx = fmaxf(mx, __shfl_xor(mx, 4, 32));
            mx = fmaxf(mx, __shfl_xor(mx, 8, 32));
            float mnew = fmaxf(mrow[r], mx);
            alpha[r] = __expf(mrow[r] - mnew);
            float p0 = __expf(a0 - mnew);
            float p1 = __expf(a1 - mnew);
            float rs = p0 + p1;
            rs += __shfl_xor(rs, 1, 32);
            rs += __shfl_xor(rs, 2, 32);
            rs += __shfl_xor(rs, 4, 32);
            rs += __shfl_xor(rs, 8, 32);
            lrow[r] = lrow[r] * alpha[r] + rs;
            mrow[r] = mnew;
            // spill P (D layout: VGPR r -> rows r / r+8, lane -> column)
            int row = r + kh;
            Plds[wave][row][lane & 15]        = f2bf(p0);
            Plds[wave][row][16 + (lane & 15)] = f2bf(p1);
        }
#pragma unroll
        for (int f = 0; f < 4; ++f)
#pragma unroll
            for (int r = 0; r < 8; ++r) oacc[f][r] *= alpha[r];
        __syncthreads();

        // ---- P as A-fragment (16 rows x 32 keys)
        v16bf pa;
        {
            int m = lane & 15;
#pragma unroll
            for (int j = 0; j < 8; ++j) {
                int c = frag_kbase(j, lane);
                pa[2 * j]     = Plds[wave][m][c];
                pa[2 * j + 1] = Plds[wave][m][c + 1];
            }
        }

        // ---- O += P @ V  (V as 32x16 B-frags from transposed LDS)
#pragma unroll
        for (int f = 0; f < 4; ++f) {
            v16bf vf;
            int dvcol = f * 16 + (lane & 15);
#pragma unroll
            for (int j = 0; j < 8; ++j) {
                int kk = frag_kbase(j, lane);
                vf[2 * j]     = Vt[dvcol][kk];
                vf[2 * j + 1] = Vt[dvcol][kk + 1];
            }
            oacc[f] = wmma_bf16(pa, vf, oacc[f]);
        }
        __syncthreads();
    }

    // ---- normalize and store bf16 attention output [B*Nq, 1024]
#pragma unroll
    for (int f = 0; f < 4; ++f) {
        int col = h * 64 + f * 16 + (lane & 15);
#pragma unroll
        for (int r = 0; r < 8; ++r) {
            int row = n0 + r + kh;
            o[(size_t)(b * Nq + row) * LD + col] = f2bf(oacc[f][r] / lrow[r]);
        }
    }
}

// ---------------------------------------------------------------------------
// Launcher
//   inputs: x, y, Wq, bq, Wk, bk, Wv, bv, Wo, bo (all fp32)
//   output: fp32 [2, 2048, 1024]
// ---------------------------------------------------------------------------
extern "C" void kernel_launch(void* const* d_in, const int* in_sizes, int n_in,
                              void* d_out, int out_size, void* d_ws, size_t ws_size,
                              hipStream_t stream) {
    const float* x  = (const float*)d_in[0];
    const float* y  = (const float*)d_in[1];
    const float* Wq = (const float*)d_in[2];
    const float* bq = (const float*)d_in[3];
    const float* Wk = (const float*)d_in[4];
    const float* bk = (const float*)d_in[5];
    const float* Wv = (const float*)d_in[6];
    const float* bv = (const float*)d_in[7];
    const float* Wo = (const float*)d_in[8];
    const float* bo = (const float*)d_in[9];
    float* out = (float*)d_out;

    const int B = 2, N = 2048, C = 1024;
    const int M = B * N;                 // 4096 rows flattened

    __bf16* qb = (__bf16*)d_ws;          // [4096,1024] bf16
    __bf16* kb = qb + (size_t)M * C;
    __bf16* vb = kb + (size_t)M * C;
    __bf16* ab = vb + (size_t)M * C;     // attention output, bf16

    dim3 gg(M / 128, C / 64);            // 32 x 16 blocks, 8 waves each

    // Q/K/V projections (fp32 A -> bf16 out)
    gemm_bias_kernel<false, true, false><<<gg, 256, 0, stream>>>(x, Wq, bq, qb, M, C, C);
    gemm_bias_kernel<false, true, false><<<gg, 256, 0, stream>>>(y, Wk, bk, kb, M, C, C);
    gemm_bias_kernel<false, true, false><<<gg, 256, 0, stream>>>(y, Wv, bv, vb, M, C, C);

    // attention: grid (64-row tiles, heads, batch), 4 waves per block
    const float scale = 0.022097086912079608f;   // 1/sqrt(2048) per reference
    flash_attn_kernel<<<dim3(N / 64, 16, B), 128, 0, stream>>>(qb, kb, vb, ab, N, N, scale);

    // output projection + ReLU (bf16 A -> fp32 out)
    gemm_bias_kernel<true, false, true><<<gg, 256, 0, stream>>>(ab, Wo, bo, out, M, C, C);
}
